// clique_guided_policy_net_61168924230077
// MI455X (gfx1250) — matrix operand order repaired and checked by reading, not compiled, not run on hardware
//
#include <hip/hip_runtime.h>
#include <math.h>

// ---------------- problem constants (match reference dims) ----------------
#define N_    8192
#define KNB   15
#define E_    (N_*KNB)
#define FT_   20
#define HGd   128
#define HLd   128
#define DEd   64
#define ATTd  16
#define ZDd   32
#define HEd   128
#define DT_   0.1f

typedef __attribute__((ext_vector_type(16))) _Float16 v16h;
typedef __attribute__((ext_vector_type(8)))  _Float16 h8;
typedef __attribute__((ext_vector_type(4)))  _Float16 h4;
typedef __attribute__((ext_vector_type(8)))  float    v8f;
typedef __attribute__((ext_vector_type(4)))  float    f4;

__device__ __forceinline__ float sigm(float x) { return 1.f / (1.f + expf(-x)); }

// =====================================================================
// WMMA GEMM:  C[M,Nc] = act( A[M,KP] @ W[Npad,KP]^T + bias (+C) )
//   - A rows are zero-padded to KP (multiple of 32): NO guards in K-loop
//   - W is f16 [Npad(out,x16), KP(in,x32)], zero-padded rows/cols
//   - each wave computes a 32x16 tile (2 accumulators, shared B fragment)
//   - M must be a multiple of 32 (true for all call sites: 8192, 122880)
//   flags: bit0 = relu, bit1 = accumulate into C
// =====================================================================
__global__ __launch_bounds__(256)
void gemm_wmma_k(const float* __restrict__ A, const _Float16* __restrict__ W,
                 const float* __restrict__ bias, float* __restrict__ C,
                 int M, int Nc, int Npad, int KP, int flags)
{
    const int lane = threadIdx.x & 31;
    const int wid  = blockIdx.x * (blockDim.x >> 5) + (threadIdx.x >> 5);
    const int tnc  = Npad >> 4;
    const int tmc  = M >> 5;                 // 32 rows per wave
    if (wid >= tmc * tnc) return;
    const int tm  = wid / tnc;
    const int tn  = wid - tm * tnc;
    const int g   = lane >> 4;               // lane group (0/1)
    const int l16 = lane & 15;
    const int row0 = tm * 32 + l16;
    const int col  = tn * 16 + l16;
    const float*    Ar0 = A + (long long)row0 * KP;
    const float*    Ar1 = Ar0 + (long long)16 * KP;
    const _Float16* Wr  = W + (long long)col * KP;

    v8f acc0 = {}, acc1 = {};
    for (int k0 = 0; k0 < KP; k0 += 32) {
        // A fragment halves: i=0..7 -> K = k0+8g+i ; i=8..15 -> K = k0+16+8g+(i-8)
        const f4* pa0  = (const f4*)(Ar0 + k0 + 8 * g);
        const f4* pa0b = (const f4*)(Ar0 + k0 + 16 + 8 * g);
        const f4* pa1  = (const f4*)(Ar1 + k0 + 8 * g);
        const f4* pa1b = (const f4*)(Ar1 + k0 + 16 + 8 * g);
        // B fragment halves: i=0..7 -> K = k0+8g+i ; i=8..15 -> K = k0+16+8g+(i-8)
        const h8* pb   = (const h8*)(Wr + k0 + 8 * g);
        const h8* pbb  = (const h8*)(Wr + k0 + 16 + 8 * g);

        h4 a00 = __builtin_convertvector(pa0[0],  h4);
        h4 a01 = __builtin_convertvector(pa0[1],  h4);
        h4 a02 = __builtin_convertvector(pa0b[0], h4);
        h4 a03 = __builtin_convertvector(pa0b[1], h4);
        h4 a10 = __builtin_convertvector(pa1[0],  h4);
        h4 a11 = __builtin_convertvector(pa1[1],  h4);
        h4 a12 = __builtin_convertvector(pa1b[0], h4);
        h4 a13 = __builtin_convertvector(pa1b[1], h4);
        h8 b0 = pb[0], b1 = pbb[0];

        v16h af0, af1, bf;
#pragma unroll
        for (int i = 0; i < 4; ++i) {
            af0[i]      = a00[i]; af0[4 + i]  = a01[i];
            af0[8 + i]  = a02[i]; af0[12 + i] = a03[i];
            af1[i]      = a10[i]; af1[4 + i]  = a11[i];
            af1[8 + i]  = a12[i]; af1[12 + i] = a13[i];
        }
#pragma unroll
        for (int i = 0; i < 8; ++i) { bf[i] = b0[i]; bf[8 + i] = b1[i]; }

        acc0 = __builtin_amdgcn_wmma_f32_16x16x32_f16(false, af0, false, bf,
                                                      (short)0, acc0, false, false);
        acc1 = __builtin_amdgcn_wmma_f32_16x16x32_f16(false, af1, false, bf,
                                                      (short)0, acc1, false, false);
    }
    if (col >= Nc) return;                    // only for Nc=2 layers
    float bv = bias ? bias[col] : 0.f;
#pragma unroll
    for (int r = 0; r < 8; ++r) {
        int m0 = tm * 32 + g * 8 + r;         // C layout: vgpr r, group g
        long long o0 = (long long)m0 * Nc + col;
        long long o1 = o0 + (long long)16 * Nc;
        float v0 = acc0[r] + bv, v1 = acc1[r] + bv;
        if (flags & 2) { v0 += C[o0]; v1 += C[o1]; }
        if (flags & 1) { v0 = fmaxf(v0, 0.f); v1 = fmaxf(v1, 0.f); }
        C[o0] = v0; C[o1] = v1;
    }
}

// ------- weight f32 -> f16 [outPad, KP] with zero padding (opt transpose) --
__global__ void convw_k(const float* __restrict__ src, _Float16* __restrict__ dst,
                        int outN, int outPad, int K, int KP, int trans)
{
    int i = blockIdx.x * blockDim.x + threadIdx.x;
    if (i >= outPad * KP) return;
    int r = i / KP, k = i - r * KP;
    float v = 0.f;
    if (r < outN && k < K)
        v = trans ? src[(long long)k * outN + r] : src[(long long)r * K + k];
    dst[i] = (_Float16)v;
}

// -------- init: prev=bsh[-1], prev_st=bsh_st[-1], pdes=0, scalars=0,
//          hstp = pad32(bsh_st[-1]) -------------------------------------
__global__ void init_k(const float* __restrict__ bsh, const float* __restrict__ bsh_st,
                       float* __restrict__ prev, float* __restrict__ pst,
                       float* __restrict__ pdes, float* __restrict__ hstp,
                       float* __restrict__ scal)
{
    int n = blockIdx.x * blockDim.x + threadIdx.x;
    if (n < N_) {
#pragma unroll
        for (int c = 0; c < 4; ++c) {
            float a = bsh   [(7LL * N_ + n) * 4 + c];
            float b = bsh_st[(7LL * N_ + n) * 4 + c];
            prev[n * 4 + c] = a;
            pst [n * 4 + c] = b;
            hstp[n * 32 + c] = b;
        }
#pragma unroll
        for (int c = 4; c < 32; ++c) hstp[n * 32 + c] = 0.f;
        pdes[n * 2] = 0.f; pdes[n * 2 + 1] = 0.f;
    }
    if (blockIdx.x == 0 && threadIdx.x < 4) scal[threadIdx.x] = 0.f;
}

// ---------------- xz = concat(nhe[-1], bz) ---------------------------------
__global__ void xz_k(const float* __restrict__ nhe_last, const float* __restrict__ bz,
                     float* __restrict__ xz)
{
    int i = blockIdx.x * blockDim.x + threadIdx.x;
    if (i >= N_ * (HEd + ZDd)) return;
    int n = i / (HEd + ZDd), j = i - n * (HEd + ZDd);
    xz[i] = (j < HEd) ? nhe_last[(long long)n * HEd + j] : bz[n * ZDd + (j - HEd)];
}

// ---------------- guide GRU pointwise update -------------------------------
__global__ void gru_k(float* __restrict__ hg, const float* __restrict__ gi,
                      const float* __restrict__ gh)
{
    int i = blockIdx.x * blockDim.x + threadIdx.x;
    if (i >= N_ * HGd) return;
    int n = i >> 7, j = i & 127;
    const float* gin = gi + (long long)n * 3 * HGd;
    const float* ghn = gh + (long long)n * 3 * HGd;
    float r  = sigm(gin[j]            + ghn[j]);
    float z  = sigm(gin[HGd + j]      + ghn[HGd + j]);
    float nn = tanhf(gin[2 * HGd + j] + r * ghn[2 * HGd + j]);
    hg[i] = (1.f - z) * nn + z * hg[i];
}

// -------- edge gather + collision + edge-MLP input assembly (stride 32) ----
__global__ void edge_k(const float* __restrict__ prev, const float* __restrict__ bns,
                       const int* __restrict__ i1, const int* __restrict__ i2,
                       float* __restrict__ Ein, float* __restrict__ collAcc)
{
    int e = blockIdx.x * blockDim.x + threadIdx.x;
    float sp = 0.f;
    if (e < E_) {
        int a = i1[e], b = i2[e];
        const float* s1 = prev + (long long)a * 4;
        const float* s2 = prev + (long long)b * 4;
        float* o = Ein + (long long)e * 32;
        o[0] = s1[0]; o[1] = s1[1]; o[2] = s1[2]; o[3] = s1[3];
        o[4] = s2[0]; o[5] = s2[1]; o[6] = s2[2]; o[7] = s2[3];
        float a0 = bns[a * 2], a1 = bns[a * 2 + 1];
        float b0 = bns[b * 2], b1 = bns[b * 2 + 1];
        o[8] = a0; o[9] = a1; o[10] = b0; o[11] = b1;
#pragma unroll
        for (int j = 12; j < 32; ++j) o[j] = 0.f;
        float dx = s1[0] - s2[0], dy = s1[1] - s2[1];
        float dist = sqrtf(dx * dx + dy * dy + 1e-9f);
        float x = 0.5f * (a0 + a1 + b0 + b1) - dist;   // rad - dist
        sp = (x > 20.f) ? x : log1pf(expf(x));         // softplus
    }
    for (int m = 16; m >= 1; m >>= 1) sp += __shfl_xor(sp, m);
    if ((threadIdx.x & 31) == 0) atomicAdd(collAcc, sp);
}

// -------- rel (padded to 32) / next_wp + tracking sum-of-squares -----------
__global__ void rel_k(const float* __restrict__ pst, const float* __restrict__ hist_st,
                      const float* __restrict__ pdes, const float* __restrict__ des_t,
                      int t, float* __restrict__ relp, float* __restrict__ nwp,
                      float* __restrict__ sumsq)
{
    int n = blockIdx.x * blockDim.x + threadIdx.x;
    float ss = 0.f;
    if (n < N_) {
        float r0, r1, r2, r3;
        if (t == 0) {
            r0 = hist_st[n * 4];     r1 = hist_st[n * 4 + 1];
            r2 = hist_st[n * 4 + 2]; r3 = hist_st[n * 4 + 3];
        } else {
            r0 = pst[n * 4]     - hist_st[n * 4]     - pdes[n * 2];
            r1 = pst[n * 4 + 1] - hist_st[n * 4 + 1] - pdes[n * 2 + 1];
            r2 = pst[n * 4 + 2] - hist_st[n * 4 + 2];
            r3 = pst[n * 4 + 3] - hist_st[n * 4 + 3];
        }
        float* rp = relp + (long long)n * 32;
        rp[0] = r0; rp[1] = r1; rp[2] = r2; rp[3] = r3;
#pragma unroll
        for (int j = 4; j < 32; ++j) rp[j] = 0.f;
        nwp[n * 2]     = des_t[n * 2]     - pst[n * 4];
        nwp[n * 2 + 1] = des_t[n * 2 + 1] - pst[n * 4 + 1];
        ss = r0 * r0 + r1 * r1;
    }
    for (int m = 16; m >= 1; m >>= 1) ss += __shfl_xor(ss, m);
    if ((threadIdx.x & 31) == 0) atomicAdd(sumsq, ss);
}

__global__ void track_k(float* __restrict__ scal)
{   // scal[0]=track, scal[1]=coll, scal[2]=sumsq
    if (threadIdx.x == 0 && blockIdx.x == 0) {
        scal[0] += sqrtf(scal[2]) / (float)N_;
        scal[2] = 0.f;
    }
}

// -------- additive attention over K=15 neighbors (1 wave per node) ---------
__global__ __launch_bounds__(256)
void attn_k(const float* __restrict__ eatt, const float* __restrict__ enc,
            const float* __restrict__ reld, const int* __restrict__ nobs,
            const float* __restrict__ b_att, const float* __restrict__ v_att,
            float* __restrict__ ctx)
{
    int lane = threadIdx.x & 31;
    int n = blockIdx.x * (blockDim.x >> 5) + (threadIdx.x >> 5);
    if (n >= N_) return;
    int idx = 0; float score = -1e30f;
    if (lane < KNB) {
        idx = nobs[n * KNB + lane];
        const float* rd = reld + (long long)n * ATTd;
        const float* ea = (idx > 0) ? (eatt + (long long)(idx - 1) * ATTd) : nullptr;
        float s = 0.f;
#pragma unroll
        for (int j = 0; j < ATTd; ++j) {
            float e = ea ? ea[j] : 0.f;
            s += v_att[j] * tanhf(e + rd[j] + b_att[j]);
        }
        score = s;
    }
    float mx = score;
    for (int m = 16; m >= 1; m >>= 1) mx = fmaxf(mx, __shfl_xor(mx, m));
    float ex = (lane < KNB) ? expf(score - mx) : 0.f;
    float sm = ex;
    for (int m = 16; m >= 1; m >>= 1) sm += __shfl_xor(sm, m);
    float w = ex / sm;
    float c0 = 0.f, c1 = 0.f;                 // each lane owns 2 of 64 channels
#pragma unroll
    for (int k = 0; k < KNB; ++k) {
        float wk = __shfl(w, k);
        int   ik = __shfl(idx, k);
        if (ik > 0) {
            const float* er = enc + (long long)(ik - 1) * DEd;
            c0 += wk * er[lane * 2];
            c1 += wk * er[lane * 2 + 1];
        }
    }
    ctx[(long long)n * DEd + lane * 2]     = c0;
    ctx[(long long)n * DEd + lane * 2 + 1] = c1;
}

// ---------------- LSTM gates (PyTorch i,f,g,o order) -----------------------
__global__ void lstm_k(const float* __restrict__ g, float* __restrict__ h,
                       float* __restrict__ c)
{
    int i = blockIdx.x * blockDim.x + threadIdx.x;
    if (i >= N_ * HLd) return;
    int n = i >> 7, j = i & 127;
    const float* gn = g + (long long)n * 4 * HLd;
    float i_ = gn[j], f_ = gn[HLd + j], gg = gn[2 * HLd + j], o_ = gn[3 * HLd + j];
    float cc = sigm(f_) * c[i] + sigm(i_) * tanhf(gg);
    c[i] = cc;
    h[i] = sigm(o_) * tanhf(cc);
}

// ------- a_in = concat(rel, next_wp, bns, h, bz) padded to 192 -------------
__global__ void ain_k(const float* __restrict__ relp, const float* __restrict__ nwp,
                      const float* __restrict__ bns, const float* __restrict__ h,
                      const float* __restrict__ bz, float* __restrict__ ain)
{
    int i = blockIdx.x * blockDim.x + threadIdx.x;
    if (i >= N_ * 192) return;
    int n = i / 192, j = i - n * 192;
    float v;
    if      (j < 4)   v = relp[(long long)n * 32 + j];
    else if (j < 6)   v = nwp[n * 2 + j - 4];
    else if (j < 8)   v = bns[n * 2 + j - 6];
    else if (j < 136) v = h[(long long)n * HLd + j - 8];
    else if (j < 168) v = bz[n * ZDd + j - 136];
    else              v = 0.f;
    ain[i] = v;
}

// ---------------- dynamics integrate + output write ------------------------
__global__ void dyn_k(float* __restrict__ prev, float* __restrict__ pst,
                      float* __restrict__ pdes, const float* __restrict__ u,
                      const float* __restrict__ des_t, float* __restrict__ out, int t)
{
    int n = blockIdx.x * blockDim.x + threadIdx.x;
    if (n >= N_) return;
    float p0 = prev[n * 4], p1 = prev[n * 4 + 1], p2 = prev[n * 4 + 2], p3 = prev[n * 4 + 3];
    float u0 = u[n * 2], u1 = u[n * 2 + 1];
    float n0 = p0 + DT_ * p2, n1 = p1 + DT_ * p3;
    float n2 = p2 + DT_ * u0, n3 = p3 + DT_ * u1;
    prev[n * 4] = n0; prev[n * 4 + 1] = n1; prev[n * 4 + 2] = n2; prev[n * 4 + 3] = n3;
    pst [n * 4] = n0; pst [n * 4 + 1] = n1; pst [n * 4 + 2] = n2; pst [n * 4 + 3] = n3;
    pdes[n * 2] = des_t[n * 2]; pdes[n * 2 + 1] = des_t[n * 2 + 1];
    float* o = out + ((long long)t * N_ + n) * 7;
    o[0] = n0; o[1] = n1; o[2] = n2; o[3] = n3; o[4] = u0; o[5] = u1;
}

// ---------------- final cost broadcast -------------------------------------
__global__ void cost_k(const float* __restrict__ scal, float* __restrict__ out)
{
    long long i = (long long)blockIdx.x * blockDim.x + threadIdx.x;
    if (i < (long long)FT_ * N_) out[i * 7 + 6] = scal[0] + scal[1];
}

// =====================================================================
// workspace layout (float units)
// =====================================================================
#define OFF_SCAL 0LL
#define OFF_DES  16LL
#define OFF_HG   (OFF_DES + (long long)FT_*N_*2)
#define OFF_GI   (OFF_HG  + (long long)N_*HGd)
#define OFF_GH   (OFF_GI  + (long long)N_*3*HGd)
#define OFF_XZ   (OFF_GH  + (long long)N_*3*HGd)
#define OFF_H    (OFF_XZ  + (long long)N_*(HEd+ZDd))
#define OFF_C    (OFF_H   + (long long)N_*HLd)
#define OFF_PREV (OFF_C   + (long long)N_*HLd)
#define OFF_PST  (OFF_PREV+ (long long)N_*4)
#define OFF_PDES (OFF_PST + (long long)N_*4)
#define OFF_HSTP (OFF_PDES+ (long long)N_*2)
#define OFF_EIN  (OFF_HSTP+ (long long)N_*32)
#define OFF_EH   (OFF_EIN + (long long)E_*32)
#define OFF_ENC  (OFF_EH  + (long long)E_*DEd)
#define OFF_EATT (OFF_ENC + (long long)E_*DEd)
#define OFF_RELP (OFF_EATT+ (long long)E_*ATTd)
#define OFF_NWP  (OFF_RELP+ (long long)N_*32)
#define OFF_RELD (OFF_NWP + (long long)N_*2)
#define OFF_CTX  (OFF_RELD+ (long long)N_*ATTd)
#define OFF_G    (OFF_CTX + (long long)N_*DEd)
#define OFF_AIN  (OFF_G   + (long long)N_*4*HLd)
#define OFF_FC1  (OFF_AIN + (long long)N_*192)
#define OFF_U    (OFF_FC1 + (long long)N_*128)
#define OFF_F16  (OFF_U   + (long long)N_*2)

// f16 weight offsets (half units, relative to f16 base); sizes outPad*KP
#define H_WH0   0LL
#define H_WC0   (H_WH0   + 128LL*32)
#define H_WGHID (H_WC0   + 128LL*32)
#define H_WGIH  (H_WGHID + 128LL*160)
#define H_WGHH  (H_WGIH  + 384LL*160)
#define H_WPROJ (H_WGHH  + 384LL*128)
#define H_WATTE (H_WPROJ + 16LL*128)
#define H_WATTD (H_WATTE + 16LL*64)
#define H_WLIH  (H_WATTD + 16LL*32)
#define H_WLHH  (H_WLIH  + 512LL*64)
#define H_WA1   (H_WLHH  + 512LL*128)
#define H_WA2   (H_WA1   + 128LL*192)
#define H_WE1   (H_WA2   + 16LL*128)
#define H_WE2   (H_WE1   + 64LL*32)

static inline void launch_gemm(hipStream_t s, const float* A, const _Float16* W,
                               const float* b, float* C, int M, int Nc, int KP, int flags)
{
    int npad   = (Nc + 15) & ~15;
    int tiles  = (M >> 5) * (npad >> 4);
    int blocks = (tiles + 7) / 8;            // 8 waves (256 threads) per block
    gemm_wmma_k<<<blocks, 256, 0, s>>>(A, W, b, C, M, Nc, npad, KP, flags);
}
static inline void launch_conv(hipStream_t s, const float* src, _Float16* dst,
                               int outN, int outPad, int K, int KP, int trans)
{
    int n = outPad * KP;
    convw_k<<<(n + 255) / 256, 256, 0, s>>>(src, dst, outN, outPad, K, KP, trans);
}

extern "C" void kernel_launch(void* const* d_in, const int* in_sizes, int n_in,
                              void* d_out, int out_size, void* d_ws, size_t ws_size,
                              hipStream_t stream)
{
    (void)in_sizes; (void)n_in; (void)out_size; (void)ws_size;
    // ---- inputs (setup_inputs order) ----
    const float* bsh     = (const float*)d_in[0];
    const float* bsh_st  = (const float*)d_in[1];
    const float* nhe     = (const float*)d_in[2];
    const float* bns     = (const float*)d_in[3];
    const float* bz      = (const float*)d_in[4];
    const int*   ei1     = (const int*)d_in[5];
    const int*   ei2     = (const int*)d_in[6];
    const int*   nobs    = (const int*)d_in[7];
    // d_in[8] = ft (compile-time FT_=20 per reference dims)
    const float* W_h0    = (const float*)d_in[9];
    const float* b_h0    = (const float*)d_in[10];
    const float* W_c0    = (const float*)d_in[11];
    const float* b_c0    = (const float*)d_in[12];
    const float* W_ghid  = (const float*)d_in[13];
    const float* b_ghid  = (const float*)d_in[14];
    const float* W_gih   = (const float*)d_in[15];
    const float* b_gih   = (const float*)d_in[16];
    const float* W_ghh   = (const float*)d_in[17];
    const float* b_ghh   = (const float*)d_in[18];
    const float* W_proj  = (const float*)d_in[19];
    const float* b_proj  = (const float*)d_in[20];
    const float* W_atte  = (const float*)d_in[21];
    const float* W_attd  = (const float*)d_in[22];
    const float* b_att   = (const float*)d_in[23];
    const float* v_att   = (const float*)d_in[24];
    const float* W_lih   = (const float*)d_in[25];
    const float* b_lih   = (const float*)d_in[26];
    const float* W_lhh   = (const float*)d_in[27];
    const float* b_lhh   = (const float*)d_in[28];
    const float* W_a1    = (const float*)d_in[29];
    const float* b_a1    = (const float*)d_in[30];
    const float* W_a2    = (const float*)d_in[31];
    const float* b_a2    = (const float*)d_in[32];
    const float* W_e1    = (const float*)d_in[33];
    const float* b_e1    = (const float*)d_in[34];
    const float* W_e2    = (const float*)d_in[35];
    const float* b_e2    = (const float*)d_in[36];

    float* ws  = (float*)d_ws;
    float* out = (float*)d_out;
    _Float16* wh = (_Float16*)(ws + OFF_F16);

    float* scal = ws + OFF_SCAL;       // [0]=track [1]=coll [2]=sumsq
    float* des  = ws + OFF_DES;
    float* hg   = ws + OFF_HG;
    float* gi   = ws + OFF_GI;
    float* gh   = ws + OFF_GH;
    float* xz   = ws + OFF_XZ;
    float* h    = ws + OFF_H;
    float* c    = ws + OFF_C;
    float* prev = ws + OFF_PREV;
    float* pst  = ws + OFF_PST;
    float* pdes = ws + OFF_PDES;
    float* hstp = ws + OFF_HSTP;
    float* Ein  = ws + OFF_EIN;
    float* eh   = ws + OFF_EH;
    float* enc  = ws + OFF_ENC;
    float* eatt = ws + OFF_EATT;
    float* relp = ws + OFF_RELP;
    float* nwp  = ws + OFF_NWP;
    float* reld = ws + OFF_RELD;
    float* ctx  = ws + OFF_CTX;
    float* g    = ws + OFF_G;
    float* ain  = ws + OFF_AIN;
    float* fc1  = ws + OFF_FC1;
    float* u    = ws + OFF_U;

    // ---- one-time: convert weights to padded f16 [outPad, KP] ----
    launch_conv(stream, W_h0,   wh + H_WH0,   128, 128, 4,   32,  0);
    launch_conv(stream, W_c0,   wh + H_WC0,   128, 128, 4,   32,  0);
    launch_conv(stream, W_ghid, wh + H_WGHID, 128, 128, 160, 160, 0);
    launch_conv(stream, W_gih,  wh + H_WGIH,  384, 384, 160, 160, 0);
    launch_conv(stream, W_ghh,  wh + H_WGHH,  384, 384, 128, 128, 0);
    launch_conv(stream, W_proj, wh + H_WPROJ, 2,   16,  128, 128, 0);
    launch_conv(stream, W_atte, wh + H_WATTE, 16,  16,  64,  64,  1); // [64,16]->[16,64]
    launch_conv(stream, W_attd, wh + H_WATTD, 16,  16,  4,   32,  1); // [4,16] ->[16,4]
    launch_conv(stream, W_lih,  wh + H_WLIH,  512, 512, 64,  64,  0);
    launch_conv(stream, W_lhh,  wh + H_WLHH,  512, 512, 128, 128, 0);
    launch_conv(stream, W_a1,   wh + H_WA1,   128, 128, 168, 192, 0);
    launch_conv(stream, W_a2,   wh + H_WA2,   2,   16,  128, 128, 0);
    launch_conv(stream, W_e1,   wh + H_WE1,   64,  64,  12,  32,  0);
    launch_conv(stream, W_e2,   wh + H_WE2,   64,  64,  64,  64,  0);

    // ---- init state ----
    init_k<<<(N_ + 255) / 256, 256, 0, stream>>>(bsh, bsh_st, prev, pst, pdes, hstp, scal);

    // ---- guide GRU: des_traj ----
    xz_k<<<(N_ * (HEd + ZDd) + 255) / 256, 256, 0, stream>>>(nhe + 7LL * N_ * HEd, bz, xz);
    launch_gemm(stream, xz, wh + H_WGHID, b_ghid, hg, N_, HGd,   160, 0); // h0g
    launch_gemm(stream, xz, wh + H_WGIH,  b_gih,  gi, N_, 3*HGd, 160, 0); // gates-in
    for (int t = 0; t < FT_; ++t) {
        launch_gemm(stream, hg, wh + H_WGHH, b_ghh, gh, N_, 3*HGd, 128, 0);
        gru_k<<<(N_ * HGd + 255) / 256, 256, 0, stream>>>(hg, gi, gh);
        launch_gemm(stream, hg, wh + H_WPROJ, b_proj, des + (long long)t * N_ * 2,
                    N_, 2, 128, 0);
    }

    // ---- obs-LSTM h0/c0 ----
    launch_gemm(stream, hstp, wh + H_WH0, b_h0, h, N_, HLd, 32, 0);
    launch_gemm(stream, hstp, wh + H_WC0, b_c0, c, N_, HLd, 32, 0);

    // ---- rollout ----
    for (int t = 0; t < FT_; ++t) {
        const float* des_t = des + (long long)t * N_ * 2;
        // edges: gather + collision + edge-MLP
        edge_k<<<(E_ + 255) / 256, 256, 0, stream>>>(prev, bns, ei1, ei2, Ein, scal + 1);
        launch_gemm(stream, Ein, wh + H_WE1, b_e1, eh,  E_, DEd, 32, 1); // relu
        launch_gemm(stream, eh,  wh + H_WE2, b_e2, enc, E_, DEd, 64, 0);
        launch_gemm(stream, enc, wh + H_WATTE, nullptr, eatt, E_, ATTd, 64, 0);
        // rel / next_wp / tracking
        rel_k<<<(N_ + 255) / 256, 256, 0, stream>>>(pst, bsh_st + 7LL * N_ * 4,
                                                    pdes, des_t, t, relp, nwp, scal + 2);
        track_k<<<1, 32, 0, stream>>>(scal);
        launch_gemm(stream, relp, wh + H_WATTD, nullptr, reld, N_, ATTd, 32, 0);
        // attention -> ctx
        attn_k<<<N_ / 8, 256, 0, stream>>>(eatt, enc, reld, nobs, b_att, v_att, ctx);
        // LSTM
        launch_gemm(stream, ctx, wh + H_WLIH, b_lih, g, N_, 4*HLd, 64,  0);
        launch_gemm(stream, h,   wh + H_WLHH, b_lhh, g, N_, 4*HLd, 128, 2); // accumulate
        lstm_k<<<(N_ * HLd + 255) / 256, 256, 0, stream>>>(g, h, c);
        // action MLP + dynamics
        ain_k<<<(N_ * 192 + 255) / 256, 256, 0, stream>>>(relp, nwp, bns, h, bz, ain);
        launch_gemm(stream, ain, wh + H_WA1, b_a1, fc1, N_, 128, 192, 1); // relu
        launch_gemm(stream, fc1, wh + H_WA2, b_a2, u,   N_, 2,   128, 0);
        dyn_k<<<(N_ + 255) / 256, 256, 0, stream>>>(prev, pst, pdes, u, des_t, out, t);
    }

    // ---- broadcast cost ----
    cost_k<<<(int)(((long long)FT_ * N_ + 255) / 256), 256, 0, stream>>>(scal, out);
}
